// MambaBlock_31971736552174
// MI455X (gfx1250) — compile-verified, matrix operand order checked by reading, not run on hardware
//
#include <hip/hip_runtime.h>
#include <hip/hip_bf16.h>

// Problem constants (match reference)
#define BB    4
#define LL    2048
#define DD    1024
#define DI    2048          // D_INNER
#define NS    16            // D_STATE
#define RNK   64            // DT_RANK
#define MTOK  (BB * LL)     // 8192 tokens

typedef __attribute__((ext_vector_type(16))) __bf16 v16bf;
typedef __attribute__((ext_vector_type(8)))  __bf16 v8bf;
typedef __attribute__((ext_vector_type(8)))  float  v8f;

union BF16x16 { v16bf v; v8bf h[2]; };

// ---------------------------------------------------------------------------
// f32 -> bf16 conversion (weights)
// ---------------------------------------------------------------------------
__global__ __launch_bounds__(256)
void cvt_f32_to_bf16(const float* __restrict__ in, __bf16* __restrict__ out, int n) {
    int i = blockIdx.x * 256 + threadIdx.x;
    if (i < n) out[i] = (__bf16)in[i];
}

// ---------------------------------------------------------------------------
// LayerNorm over D=1024, one block per token, emit bf16 (GEMM A operand)
// ---------------------------------------------------------------------------
__global__ __launch_bounds__(256)
void layernorm_to_bf16(const float* __restrict__ x, const float* __restrict__ g,
                       const float* __restrict__ b, __bf16* __restrict__ out) {
    __shared__ float s1[256], s2[256];
    const int tok = blockIdx.x;
    const float* row = x + (size_t)tok * DD;
    float sum = 0.f, sq = 0.f;
    for (int i = threadIdx.x; i < DD; i += 256) {
        float v = row[i]; sum += v; sq += v * v;
    }
    s1[threadIdx.x] = sum; s2[threadIdx.x] = sq;
    __syncthreads();
    for (int off = 128; off > 0; off >>= 1) {
        if (threadIdx.x < off) {
            s1[threadIdx.x] += s1[threadIdx.x + off];
            s2[threadIdx.x] += s2[threadIdx.x + off];
        }
        __syncthreads();
    }
    const float mean = s1[0] * (1.f / DD);
    const float var  = s2[0] * (1.f / DD) - mean * mean;
    const float rstd = rsqrtf(var + 1e-6f);
    __bf16* orow = out + (size_t)tok * DD;
    for (int i = threadIdx.x; i < DD; i += 256)
        orow[i] = (__bf16)((row[i] - mean) * rstd * g[i] + b[i]);
}

// ---------------------------------------------------------------------------
// WMMA GEMM: C[M,N] = A[M,K](bf16) * W[N,K]^T(bf16) (+ residual), f32 out.
// Wave tile: 32x64 (8 accumulators; 2 A frags x 4 B frags, B reused for both
// M halves -> 6 fragment loads per 8 WMMAs).
// Block: 8 waves as 2(M) x 4(N) -> 64x256 of C. grid = (N/256, M/64).
// Per CDNA5 ISA 16-bit A layout:
//   lane&15 = row, lane>=16 -> K offset +8; VGPR0-3 = K[0..7], VGPR4-7 = K[16..23].
// ---------------------------------------------------------------------------
__global__ __launch_bounds__(256)
void wmma_gemm_bf16(const __bf16* __restrict__ A, const __bf16* __restrict__ W,
                    float* __restrict__ C, const float* __restrict__ Res,
                    int M, int N, int K) {
    const int lane  = threadIdx.x & 31;
    const int wave  = threadIdx.x >> 5;
    const int m0    = blockIdx.y * 64 + (wave >> 2) * 32;
    const int n0    = blockIdx.x * 256 + (wave & 3) * 64;
    const int row   = lane & 15;
    const int khalf = (lane >> 4) << 3;   // 0 or 8

    const __bf16* ap0 = A + (size_t)(m0 + row) * K + khalf;
    const __bf16* ap1 = A + (size_t)(m0 + 16 + row) * K + khalf;
    const __bf16* wp[4];
#pragma unroll
    for (int s = 0; s < 4; ++s)
        wp[s] = W + (size_t)(n0 + s * 16 + row) * K + khalf;

    v8f acc0[4] = {};   // rows m0..m0+15
    v8f acc1[4] = {};   // rows m0+16..m0+31
    for (int k0 = 0; k0 < K; k0 += 32) {
        __builtin_prefetch(ap0 + k0 + 128, 0, 1);   // global_prefetch_b8
        __builtin_prefetch(ap1 + k0 + 128, 0, 1);
        BF16x16 a0, a1;
        a0.h[0] = *(const v8bf*)(ap0 + k0);
        a0.h[1] = *(const v8bf*)(ap0 + k0 + 16);
        a1.h[0] = *(const v8bf*)(ap1 + k0);
        a1.h[1] = *(const v8bf*)(ap1 + k0 + 16);
#pragma unroll
        for (int s = 0; s < 4; ++s) {
            BF16x16 bb;
            bb.h[0] = *(const v8bf*)(wp[s] + k0);
            bb.h[1] = *(const v8bf*)(wp[s] + k0 + 16);
            acc0[s] = __builtin_amdgcn_wmma_f32_16x16x32_bf16(
                false, a0.v, false, bb.v, (short)0, acc0[s], false, false);
            acc1[s] = __builtin_amdgcn_wmma_f32_16x16x32_bf16(
                false, a1.v, false, bb.v, (short)0, acc1[s], false, false);
        }
    }

    // C/D layout: VGPR r -> M = r (lanes 0-15) / r+8 (lanes 16-31), N = lane&15
    const int ncol  = lane & 15;
    const int moff  = (lane >> 4) << 3;
#pragma unroll
    for (int s = 0; s < 4; ++s) {
        const int nc = n0 + s * 16 + ncol;
#pragma unroll
        for (int r = 0; r < 8; ++r) {
            size_t idx0 = (size_t)(m0 + moff + r) * N + nc;
            size_t idx1 = (size_t)(m0 + 16 + moff + r) * N + nc;
            float v0 = acc0[s][r];
            float v1 = acc1[s][r];
            if (Res) { v0 += Res[idx0]; v1 += Res[idx1]; }
            C[idx0] = v0;
            C[idx1] = v1;
        }
    }
}

// ---------------------------------------------------------------------------
// Causal depthwise conv1d (k=4) over time on xc half of xz, + bias + SiLU
// ---------------------------------------------------------------------------
__global__ __launch_bounds__(256)
void conv_silu(const float* __restrict__ xz, const float* __restrict__ cw,
               const float* __restrict__ cb, float* __restrict__ xc) {
    int idx = blockIdx.x * 256 + threadIdx.x;
    if (idx >= BB * LL * DI) return;
    const int d = idx % DI;
    const int l = (idx / DI) % LL;
    const int b = idx / (DI * LL);
    const size_t base = (size_t)b * LL * (2 * DI) + d;
    float acc = cb[d];
    const float w0 = cw[d * 4 + 0], w1 = cw[d * 4 + 1];
    const float w2 = cw[d * 4 + 2], w3 = cw[d * 4 + 3];
    if (l >= 3) acc += xz[base + (size_t)(l - 3) * (2 * DI)] * w0;
    if (l >= 2) acc += xz[base + (size_t)(l - 2) * (2 * DI)] * w1;
    if (l >= 1) acc += xz[base + (size_t)(l - 1) * (2 * DI)] * w2;
    acc += xz[base + (size_t)l * (2 * DI)] * w3;
    acc = acc / (1.f + __expf(-acc));            // SiLU
    xc[idx] = acc;
}

// ---------------------------------------------------------------------------
// x_proj: x_dbl[tok, 0..95] = xc[tok,:] . x_proj_w[e,:]  (K = 2048, small N)
// One block per token; token row staged in LDS.
// ---------------------------------------------------------------------------
__global__ __launch_bounds__(128)
void xproj(const float* __restrict__ xc, const float* __restrict__ xw,
           float* __restrict__ xdbl) {
    __shared__ float sx[DI];
    const int tok = blockIdx.x;
    const float* row = xc + (size_t)tok * DI;
    for (int i = threadIdx.x; i < DI; i += 128) sx[i] = row[i];
    __syncthreads();
    if (threadIdx.x < 96) {
        const float* wr = xw + (size_t)threadIdx.x * DI;
        float acc = 0.f;
        for (int k = 0; k < DI; ++k) acc += sx[k] * wr[k];
        xdbl[tok * 96 + threadIdx.x] = acc;
    }
}

// ---------------------------------------------------------------------------
// dt_proj (K=64) + softplus -> delta[tok, d]
// ---------------------------------------------------------------------------
__global__ __launch_bounds__(256)
void dtproj_softplus(const float* __restrict__ xdbl, const float* __restrict__ dw,
                     const float* __restrict__ db, float* __restrict__ delta) {
    int idx = blockIdx.x * 256 + threadIdx.x;
    if (idx >= BB * LL * DI) return;
    const int d  = idx % DI;
    const int tok = idx / DI;
    const float* r = xdbl + (size_t)tok * 96;   // dt = cols [0,64)
    const float* w = dw + (size_t)d * RNK;
    float acc = db[d];
#pragma unroll 8
    for (int k = 0; k < RNK; ++k) acc += r[k] * w[k];
    delta[idx] = (acc > 20.f) ? acc : log1pf(__expf(acc));
}

// ---------------------------------------------------------------------------
// Selective scan: 16 lanes per (b,d) channel, one state per lane.
// Sequential over L; state.C dot via 4-step shfl_xor butterfly (within 16
// lanes, valid on wave32). Fuses D-skip, SiLU(z) gate, bf16 output (GEMM A).
// grid = (B*DI)/16 channels / 16 groups-per-block = 512 blocks of 256.
// ---------------------------------------------------------------------------
__global__ __launch_bounds__(256)
void selective_scan(const float* __restrict__ delta, const float* __restrict__ u,
                    const float* __restrict__ xdbl, const float* __restrict__ alog,
                    const float* __restrict__ dpar, const float* __restrict__ xz,
                    __bf16* __restrict__ yout) {
    const int gid = blockIdx.x * 16 + (threadIdx.x >> 4);  // channel index in [0, B*DI)
    const int n   = threadIdx.x & 15;                      // state index
    const int b   = gid / DI;
    const int d   = gid % DI;
    const float An = -__expf(alog[d * NS + n]);
    const float Dd = dpar[d];
    float s = 0.f;
    const size_t tokbase = (size_t)b * LL;
    for (int t = 0; t < LL; ++t) {
        const size_t tok = tokbase + t;
        const float dl = delta[tok * DI + d];
        const float uu = u[tok * DI + d];
        const float Bn = xdbl[tok * 96 + 64 + n];
        const float Cn = xdbl[tok * 96 + 80 + n];
        s = s * __expf(dl * An) + dl * uu * Bn;
        float p = s * Cn;
        p += __shfl_xor(p, 1);
        p += __shfl_xor(p, 2);
        p += __shfl_xor(p, 4);
        p += __shfl_xor(p, 8);
        if (n == 0) {
            float y = p + uu * Dd;
            const float z = xz[tok * (2 * DI) + DI + d];
            y *= z / (1.f + __expf(-z));                   // * SiLU(z)
            yout[tok * DI + d] = (__bf16)y;
        }
    }
}

// ---------------------------------------------------------------------------
// Launch: needs ~335 MB workspace.
// ---------------------------------------------------------------------------
extern "C" void kernel_launch(void* const* d_in, const int* in_sizes, int n_in,
                              void* d_out, int out_size, void* d_ws, size_t ws_size,
                              hipStream_t stream) {
    (void)in_sizes; (void)n_in; (void)out_size; (void)ws_size;
    const float* inp  = (const float*)d_in[0];
    const float* ln_g = (const float*)d_in[1];
    const float* ln_b = (const float*)d_in[2];
    const float* win  = (const float*)d_in[3];   // (4096, 1024)
    const float* cw   = (const float*)d_in[4];   // (2048, 4)
    const float* cb   = (const float*)d_in[5];
    const float* xw   = (const float*)d_in[6];   // (96, 2048)
    const float* dw   = (const float*)d_in[7];   // (2048, 64)
    const float* db   = (const float*)d_in[8];
    const float* alog = (const float*)d_in[9];   // (2048, 16)
    const float* dpar = (const float*)d_in[10];
    const float* wout = (const float*)d_in[11];  // (1024, 2048)
    float* out = (float*)d_out;

    char* p = (char*)d_ws;
    auto carve = [&](size_t bytes) {
        char* q = p; p += (bytes + 255) & ~(size_t)255; return q;
    };
    __bf16* xn_bf   = (__bf16*)carve((size_t)MTOK * DD * 2);        // 16.8 MB
    __bf16* win_bf  = (__bf16*)carve((size_t)2 * DI * DD * 2);      //  8.4 MB
    __bf16* wout_bf = (__bf16*)carve((size_t)DD * DI * 2);          //  4.2 MB
    float*  xz      = (float*) carve((size_t)MTOK * 2 * DI * 4);    // 134  MB
    float*  xc      = (float*) carve((size_t)MTOK * DI * 4);        //  67  MB
    float*  xdbl    = (float*) carve((size_t)MTOK * 96 * 4);        //   3  MB
    float*  delta   = (float*) carve((size_t)MTOK * DI * 4);        //  67  MB
    __bf16* y_bf    = (__bf16*)carve((size_t)MTOK * DI * 2);        //  33  MB

    // weight conversions
    cvt_f32_to_bf16<<<(2 * DI * DD + 255) / 256, 256, 0, stream>>>(win, win_bf, 2 * DI * DD);
    cvt_f32_to_bf16<<<(DD * DI + 255) / 256, 256, 0, stream>>>(wout, wout_bf, DD * DI);

    // LayerNorm -> bf16
    layernorm_to_bf16<<<MTOK, 256, 0, stream>>>(inp, ln_g, ln_b, xn_bf);

    // in_proj: [8192,1024] x [4096,1024]^T -> xz[8192,4096]
    {
        dim3 g(2 * DI / 256, MTOK / 64);
        wmma_gemm_bf16<<<g, 256, 0, stream>>>(xn_bf, win_bf, xz, nullptr,
                                              MTOK, 2 * DI, DD);
    }

    // depthwise conv + SiLU
    conv_silu<<<(BB * LL * DI + 255) / 256, 256, 0, stream>>>(xz, cw, cb, xc);

    // x_proj
    xproj<<<MTOK, 128, 0, stream>>>(xc, xw, xdbl);

    // dt_proj + softplus
    dtproj_softplus<<<(BB * LL * DI + 255) / 256, 256, 0, stream>>>(xdbl, dw, db, delta);

    // selective scan (+D skip, *SiLU(z)), bf16 out
    selective_scan<<<(BB * DI) / 16, 256, 0, stream>>>(delta, xc, xdbl, alog,
                                                       dpar, xz, y_bf);

    // out_proj + residual: [8192,2048] x [1024,2048]^T + inp -> out[8192,1024]
    {
        dim3 g(DD / 256, MTOK / 64);
        wmma_gemm_bf16<<<g, 256, 0, stream>>>(y_bf, wout_bf, out, inp,
                                              MTOK, DD, DI);
    }
}
